// MultiHeadAttention_58222576664999
// MI455X (gfx1250) — matrix operand order repaired and checked by reading, not compiled, not run on hardware
//
#include <hip/hip_runtime.h>
#include <hip/hip_bf16.h>

// ---------------------------------------------------------------------------
// MHA forward for MI455X (gfx1250), wave32 + WMMA bf16 (f32 accumulate).
//   B=64, T=256, C=1024, H=16, D=64
// Pipeline: [qkv gemm] -> [flash attention per (b,h)] -> [output gemm]
// CDNA5 paths exercised:
//   - v_wmma_f32_16x16x32_bf16 (all matrix math)
//   - TENSOR_LOAD_TO_LDS (TDM DMA, TENSORcnt + s_wait_tensorcnt) for K tiles
//   - global_load_async_to_lds_b128 (ASYNCcnt + s_wait_asynccnt) for Y tiles
//   - global_prefetch_b8 for next-tile prefetch
// ---------------------------------------------------------------------------

typedef __attribute__((ext_vector_type(16))) __bf16 v16bf;
typedef __attribute__((ext_vector_type(8)))  float  v8f;
typedef __attribute__((ext_vector_type(4)))  unsigned int u32x4;
typedef __attribute__((ext_vector_type(4)))  int i32x4;
typedef __attribute__((ext_vector_type(8)))  int i32x8;

#define EMBED 1024
#define HEADS 16
#define HDIM  64
#define SEQ   256
#define BATCH 64
#define ROWS  (BATCH * SEQ)   // 16384

union FragBF {            // 16 bf16 = one WMMA A or B fragment (8 VGPRs)
    v16bf v;
    uint4 q[2];
    __bf16 h[16];
};
union FragF {             // 8 f32 = one WMMA C/D fragment
    v8f v;
    float f[8];
};

__device__ __forceinline__ v8f wmma_bf16(const FragBF& a, const FragBF& b, v8f c) {
    // D = A(16x32 bf16) * B(32x16 bf16) + C(16x16 f32)
    return __builtin_amdgcn_wmma_f32_16x16x32_bf16(
        /*neg_a=*/false, a.v, /*neg_b=*/false, b.v,
        /*c_mod=*/(short)0, c, /*reuse_a=*/false, /*reuse_b=*/false);
}

__device__ __forceinline__ v8f zero_v8f() {
    v8f z = {0.f, 0.f, 0.f, 0.f, 0.f, 0.f, 0.f, 0.f};
    return z;
}

// Async global->LDS copy of one 16B chunk per lane (GVS addressing mode).
__device__ __forceinline__ void async_copy_b128(uint32_t lds_addr,
                                                uint32_t goff,
                                                uint64_t gbase) {
    asm volatile("global_load_async_to_lds_b128 %0, %1, %2 offset:0"
                 :: "v"(lds_addr), "v"(goff), "s"(gbase) : "memory");
}
__device__ __forceinline__ void wait_async0() {
    asm volatile("s_wait_asynccnt 0" ::: "memory");
}

// TDM: DMA a 2-D row-major tile (16-bit elements) from global memory to LDS.
// D# descriptor built per CDNA5 ISA Sec 8.3/8.4:
//   group0: count=1 | lds_addr[63:32] | global_addr[120:64] | type=2[127:126]
//   group1: data_size=1 (2B) | tensor_dim0 | tensor_dim1 | tile_dim0 |
//           tile_dim1 | tensor_dim0_stride   (groups 2/3 zero: <=2-D tensor)
__device__ __forceinline__ void tdm_load_2d_to_lds(uint32_t lds_addr,
                                                   uint64_t gaddr,
                                                   uint32_t rows,
                                                   uint32_t row_elems,
                                                   uint32_t row_stride_elems) {
    u32x4 g0;
    g0.x = 1u;                                        // count = 1 (valid D#)
    g0.y = lds_addr;                                  // lds_addr  (bits 63:32)
    g0.z = (uint32_t)gaddr;                           // gaddr lo  (bits 95:64)
    g0.w = (uint32_t)(gaddr >> 32) | (2u << 30);      // gaddr hi + type=2
    i32x8 g1;
    g1[0] = (int)(1u << 16);                          // data_size=1 (2 bytes)
    g1[1] = (int)((row_elems & 0xFFFFu) << 16);       // tensor_dim0 lo16 @bit48
    g1[2] = (int)(((row_elems >> 16) & 0xFFFFu)       // tensor_dim0 hi16
                  | ((rows & 0xFFFFu) << 16));        // tensor_dim1 lo16
    g1[3] = (int)(((rows >> 16) & 0xFFFFu)            // tensor_dim1 hi16
                  | ((row_elems & 0xFFFFu) << 16));   // tile_dim0
    g1[4] = (int)(rows & 0xFFFFu);                    // tile_dim1 (tile_dim2=0)
    g1[5] = (int)row_stride_elems;                    // tensor_dim0_stride lo32
    g1[6] = 0;
    g1[7] = 0;
    i32x4 gz = {0, 0, 0, 0};
#if __clang_major__ >= 23
    i32x8 gz8 = {0, 0, 0, 0, 0, 0, 0, 0};
    __builtin_amdgcn_tensor_load_to_lds(g0, g1, gz, gz, gz8, 0);
#else
    __builtin_amdgcn_tensor_load_to_lds(g0, g1, gz, gz, 0);
#endif
}

// ---------------------------------------------------------------------------
// Kernel 1: QKV projection.  X[16384,1024] f32 @ Wqkv[1024,3072] f32 + bqkv.
// Block tile 128(M) x 64(N), K-step 32. 8 waves, each computes 32x32.
// Output scattered as bf16 into Q/K/V buffers laid out [B,H,T,D].
// (f32->bf16 conversion happens in the staging path, so no DMA copy here.)
// ---------------------------------------------------------------------------
__global__ __launch_bounds__(256) void qkv_gemm_kernel(
    const float* __restrict__ X, const float* __restrict__ W,
    const float* __restrict__ bias,
    __bf16* __restrict__ Qb, __bf16* __restrict__ Kb, __bf16* __restrict__ Vb)
{
    __shared__ __bf16 As[128 * 32];   // row-major, K contiguous     (8 KB)
    __shared__ __bf16 Bs[64 * 32];    // TRANSPOSED: Bs[n*32 + k]    (4 KB)

    const int mb   = blockIdx.x;           // 0..127
    const int nb   = blockIdx.y;           // 0..47
    const int tid  = threadIdx.x;
    const int wave = tid >> 5;
    const int lane = tid & 31;
    const int waveM = wave >> 1;           // 0..3  -> 32-row slab
    const int waveN = wave & 1;            // 0..1  -> 32-col slab
    const int lhalf = lane >> 4;
    const int lmod  = lane & 15;

    v8f acc[2][2];
    #pragma unroll
    for (int i = 0; i < 2; ++i)
        #pragma unroll
        for (int j = 0; j < 2; ++j) acc[i][j] = zero_v8f();

    for (int kb = 0; kb < EMBED / 32; ++kb) {
        // --- stage A tile (convert f32 -> bf16): thread loads 16 floats ----
        {
            const int m  = tid >> 1;
            const int k0 = (tid & 1) * 16;
            const float* src = X + (size_t)(mb * 128 + m) * EMBED + kb * 32 + k0;
            __bf16* dst = &As[m * 32 + k0];
            #pragma unroll
            for (int i = 0; i < 16; i += 4) {
                float4 f = *(const float4*)(src + i);
                dst[i + 0] = (__bf16)f.x;  dst[i + 1] = (__bf16)f.y;
                dst[i + 2] = (__bf16)f.z;  dst[i + 3] = (__bf16)f.w;
            }
        }
        // --- stage B tile transposed: Bs[n][k] = W[k][n] -------------------
        {
            const int k  = tid >> 3;
            const int n0 = (tid & 7) * 8;
            const float* src = W + (size_t)(kb * 32 + k) * (3 * EMBED) + nb * 64 + n0;
            #pragma unroll
            for (int i = 0; i < 8; i += 4) {
                float4 f = *(const float4*)(src + i);
                Bs[(n0 + i + 0) * 32 + k] = (__bf16)f.x;
                Bs[(n0 + i + 1) * 32 + k] = (__bf16)f.y;
                Bs[(n0 + i + 2) * 32 + k] = (__bf16)f.z;
                Bs[(n0 + i + 3) * 32 + k] = (__bf16)f.w;
            }
        }
        __syncthreads();

        if (kb + 1 < EMBED / 32) {   // gfx1250 global_prefetch_b8 of next tiles
            __builtin_prefetch(X + (size_t)(mb * 128 + (tid >> 1)) * EMBED + (kb + 1) * 32, 0, 0);
            __builtin_prefetch(W + (size_t)((kb + 1) * 32 + (tid >> 3)) * (3 * EMBED) + nb * 64, 0, 0);
        }

        // --- fragments (ISA 7.12.2 layouts) --------------------------------
        FragBF afr[2], bfr[2];
        #pragma unroll
        for (int mt = 0; mt < 2; ++mt) {
            const __bf16* p = &As[(waveM * 32 + mt * 16 + lmod) * 32 + lhalf * 8];
            afr[mt].q[0] = *(const uint4*)p;          // K = base .. base+7
            afr[mt].q[1] = *(const uint4*)(p + 16);   // K = base+16 .. base+23
        }
        #pragma unroll
        for (int nt = 0; nt < 2; ++nt) {
            const __bf16* p = &Bs[(waveN * 32 + nt * 16 + lmod) * 32 + lhalf * 8];
            bfr[nt].q[0] = *(const uint4*)p;          // K = half*8 .. +7
            bfr[nt].q[1] = *(const uint4*)(p + 16);   // K = 16+half*8 .. +7
        }
        #pragma unroll
        for (int mt = 0; mt < 2; ++mt)
            #pragma unroll
            for (int nt = 0; nt < 2; ++nt)
                acc[mt][nt] = wmma_bf16(afr[mt], bfr[nt], acc[mt][nt]);
        __syncthreads();
    }

    // --- epilogue: +bias, scatter bf16 into [B,H,T,D] ----------------------
    #pragma unroll
    for (int nt = 0; nt < 2; ++nt) {
        const int n = nb * 64 + waveN * 32 + nt * 16 + lmod;  // 0..3071
        const float bv = bias[n];
        const int which = n >> 10;     // 0=Q 1=K 2=V (64-col tiles never straddle)
        const int c = n & 1023;
        const int h = c >> 6, d = c & 63;
        __bf16* base = (which == 0) ? Qb : ((which == 1) ? Kb : Vb);
        #pragma unroll
        for (int mt = 0; mt < 2; ++mt) {
            FragF cf; cf.v = acc[mt][nt];
            #pragma unroll
            for (int r = 0; r < 8; ++r) {
                const int g = mb * 128 + waveM * 32 + mt * 16 + r + lhalf * 8;
                const int b = g >> 8, t = g & 255;
                base[(size_t)(((b * HEADS + h) * SEQ) + t) * HDIM + d] =
                    (__bf16)(cf.f[r] + bv);
            }
        }
    }
}

// ---------------------------------------------------------------------------
// Kernel 2: causal flash attention, one workgroup per (b,h).
// Wave w owns query rows [32w, 32w+32); loops key blocks kb=0..w (causal).
// Q fragments in registers; K staged in LDS via ONE Tensor-Data-Mover DMA
// (issued by wave 0, overlapped with the V transpose staging, completion via
// s_wait_tensorcnt); V staged transposed in LDS. Online softmax over 32-key
// blocks; P relaid via per-wave LDS staging. LDS: 32K+32K+16K = 80 KB.
// ---------------------------------------------------------------------------
__global__ __launch_bounds__(256) void attn_kernel(
    const __bf16* __restrict__ Qb, const __bf16* __restrict__ Kb,
    const __bf16* __restrict__ Vb, __bf16* __restrict__ Y)
{
    __shared__ __bf16 Ks[SEQ * HDIM];      // row-major [t][d]       (32 KB)
    __shared__ __bf16 Vt[HDIM * SEQ];      // Vt[d*256 + t]          (32 KB)
    __shared__ __bf16 Ps[8][32 * 32];      // per-wave P staging     (16 KB)

    const int bh  = blockIdx.x;            // 0..1023
    const int b   = bh >> 4, h = bh & 15;
    const int tid = threadIdx.x;
    const int wave = tid >> 5;
    const int lane = tid & 31;
    const int lhalf = lane >> 4;
    const int lmod  = lane & 15;
    const size_t head_off = (size_t)bh * SEQ * HDIM;

    // --- stage K via Tensor Data Mover (one DMA for the whole 32 KB tile) --
    if (wave == 0) {
        tdm_load_2d_to_lds((uint32_t)(uintptr_t)Ks,
                           (uint64_t)(uintptr_t)(Kb + head_off),
                           /*rows=*/SEQ, /*row_elems=*/HDIM,
                           /*row_stride_elems=*/HDIM);
    }
    // --- stage V transposed into LDS (data swizzle -> VGPR path), overlaps
    //     with the TDM transfer running in the background ------------------
    for (int i = tid; i < SEQ * HDIM / 8; i += 256) {
        const int t  = i >> 3;
        const int d0 = (i & 7) * 8;
        union { uint4 u; __bf16 hh[8]; } cv;
        cv.u = *(const uint4*)&Vb[head_off + (size_t)t * HDIM + d0];
        #pragma unroll
        for (int j = 0; j < 8; ++j) Vt[(d0 + j) * SEQ + t] = cv.hh[j];
    }
    if (wave == 0) __builtin_amdgcn_s_wait_tensorcnt(0);
    __syncthreads();

    // --- Q fragments (resident): rows 32*wave .. +32, D=64 = 2 K-chunks ----
    FragBF qf[2][2];
    #pragma unroll
    for (int mt = 0; mt < 2; ++mt)
        #pragma unroll
        for (int kc = 0; kc < 2; ++kc) {
            const __bf16* p = Qb + head_off
                + (size_t)(32 * wave + mt * 16 + lmod) * HDIM + kc * 32 + lhalf * 8;
            qf[mt][kc].q[0] = *(const uint4*)p;
            qf[mt][kc].q[1] = *(const uint4*)(p + 16);
        }

    v8f o[2][4];
    #pragma unroll
    for (int mt = 0; mt < 2; ++mt)
        #pragma unroll
        for (int dt = 0; dt < 4; ++dt) o[mt][dt] = zero_v8f();

    float m_i[2][8], l_i[2][8];
    #pragma unroll
    for (int mt = 0; mt < 2; ++mt)
        #pragma unroll
        for (int r = 0; r < 8; ++r) { m_i[mt][r] = -1e30f; l_i[mt][r] = 0.f; }

    const float scale = 0.125f;   // 1/sqrt(64)

    for (int kb = 0; kb <= wave; ++kb) {
        const bool diag = (kb == wave);

        // ---- S = Q @ K^T for 32 keys (two 16-key subtiles) ----------------
        FragF s[2][2];
        #pragma unroll
        for (int st = 0; st < 2; ++st) {
            FragBF kf[2];
            #pragma unroll
            for (int kc = 0; kc < 2; ++kc) {
                const __bf16* p =
                    &Ks[(kb * 32 + st * 16 + lmod) * HDIM + kc * 32 + lhalf * 8];
                kf[kc].q[0] = *(const uint4*)p;
                kf[kc].q[1] = *(const uint4*)(p + 16);
            }
            #pragma unroll
            for (int mt = 0; mt < 2; ++mt) {
                v8f c = zero_v8f();
                c = wmma_bf16(qf[mt][0], kf[0], c);
                c = wmma_bf16(qf[mt][1], kf[1], c);
                s[mt][st].v = c;
            }
        }

        // ---- scale, causal mask, online softmax over the 32-key block -----
        #pragma unroll
        for (int mt = 0; mt < 2; ++mt) {
            #pragma unroll
            for (int r = 0; r < 8; ++r) {
                const int qrow = 32 * wave + mt * 16 + r + lhalf * 8;
                float sv[2];
                float rowm = -1e30f;
                #pragma unroll
                for (int st = 0; st < 2; ++st) {
                    float v = s[mt][st].f[r] * scale;
                    const int krow = kb * 32 + st * 16 + lmod;
                    if (diag && krow > qrow) v = -1e30f;
                    sv[st] = v;
                    rowm = fmaxf(rowm, v);
                }
                #pragma unroll
                for (int d = 1; d < 16; d <<= 1)
                    rowm = fmaxf(rowm, __shfl_xor(rowm, d, 32));
                const float mnew  = fmaxf(m_i[mt][r], rowm);
                const float alpha = __expf(m_i[mt][r] - mnew);
                m_i[mt][r] = mnew;
                float rsum = 0.f;
                #pragma unroll
                for (int st = 0; st < 2; ++st) {
                    const float pv = __expf(sv[st] - mnew);
                    rsum += pv;
                    Ps[wave][(mt * 16 + r + lhalf * 8) * 32 + st * 16 + lmod] = (__bf16)pv;
                }
                #pragma unroll
                for (int d = 1; d < 16; d <<= 1)
                    rsum += __shfl_xor(rsum, d, 32);
                l_i[mt][r] = l_i[mt][r] * alpha + rsum;
                #pragma unroll
                for (int dt = 0; dt < 4; ++dt)
                    o[mt][dt][r] *= alpha;
            }
        }

        // ---- O += P @ V (K = 32 keys) -------------------------------------
        FragBF pf[2];
        #pragma unroll
        for (int mt = 0; mt < 2; ++mt) {
            const __bf16* p = &Ps[wave][(mt * 16 + lmod) * 32 + lhalf * 8];
            pf[mt].q[0] = *(const uint4*)p;
            pf[mt].q[1] = *(const uint4*)(p + 16);
        }
        #pragma unroll
        for (int dt = 0; dt < 4; ++dt) {
            FragBF vf;
            const __bf16* p = &Vt[(dt * 16 + lmod) * SEQ + kb * 32 + lhalf * 8];
            vf.q[0] = *(const uint4*)p;
            vf.q[1] = *(const uint4*)(p + 16);
            #pragma unroll
            for (int mt = 0; mt < 2; ++mt)
                o[mt][dt] = wmma_bf16(pf[mt], vf, o[mt][dt]);
        }
    }

    // ---- finalize: O / l, store bf16 into Y[b, t, h*64 + d] ---------------
    #pragma unroll
    for (int mt = 0; mt < 2; ++mt) {
        #pragma unroll
        for (int r = 0; r < 8; ++r) {
            const int t   = 32 * wave + mt * 16 + r + lhalf * 8;
            const float inv = 1.0f / l_i[mt][r];
            #pragma unroll
            for (int dt = 0; dt < 4; ++dt) {
                const int c = h * HDIM + dt * 16 + lmod;
                Y[(size_t)(b * SEQ + t) * EMBED + c] = (__bf16)(o[mt][dt][r] * inv);
            }
        }
    }
}

// ---------------------------------------------------------------------------
// Kernel 3: output projection. Y[16384,1024] bf16 @ Wo[1024,1024] f32 + bo
// -> f32 d_out. A tile (already bf16) staged via async global->LDS DMA.
// ---------------------------------------------------------------------------
__global__ __launch_bounds__(256) void out_gemm_kernel(
    const __bf16* __restrict__ Y, const float* __restrict__ W,
    const float* __restrict__ bias, float* __restrict__ Out)
{
    __shared__ __bf16 As[128 * 32];
    __shared__ __bf16 Bs[64 * 32];    // transposed: Bs[n*32 + k]

    const int mb   = blockIdx.x;          // 0..127
    const int nb   = blockIdx.y;          // 0..15
    const int tid  = threadIdx.x;
    const int wave = tid >> 5;
    const int lane = tid & 31;
    const int waveM = wave >> 1, waveN = wave & 1;
    const int lhalf = lane >> 4, lmod = lane & 15;

    const uint32_t as_base = (uint32_t)(uintptr_t)As;

    v8f acc[2][2];
    #pragma unroll
    for (int i = 0; i < 2; ++i)
        #pragma unroll
        for (int j = 0; j < 2; ++j) acc[i][j] = zero_v8f();

    for (int kb = 0; kb < EMBED / 32; ++kb) {
        {   // A tile: straight bf16 copy -> async global->LDS DMA (2 x b128)
            const uint64_t ysrc = (uint64_t)(uintptr_t)
                (Y + (size_t)(mb * 128) * EMBED + kb * 32);
            #pragma unroll
            for (int j = 0; j < 2; ++j) {
                const int c   = tid * 2 + j;       // 0..511 16B chunks
                const int m   = c >> 2;            // tile row
                const int p16 = c & 3;             // 16B piece within row
                const uint32_t lds_addr = as_base + (uint32_t)(m * 64 + p16 * 16);
                const uint32_t goff = (uint32_t)(m * (EMBED * 2) + p16 * 16);
                async_copy_b128(lds_addr, goff, ysrc);
            }
        }
        {   // B tile transposed + f32->bf16 (conversion -> VGPR path)
            const int k  = tid >> 3;
            const int n0 = (tid & 7) * 8;
            const float* src = W + (size_t)(kb * 32 + k) * EMBED + nb * 64 + n0;
            #pragma unroll
            for (int i = 0; i < 8; i += 4) {
                float4 f = *(const float4*)(src + i);
                Bs[(n0 + i + 0) * 32 + k] = (__bf16)f.x;
                Bs[(n0 + i + 1) * 32 + k] = (__bf16)f.y;
                Bs[(n0 + i + 2) * 32 + k] = (__bf16)f.z;
                Bs[(n0 + i + 3) * 32 + k] = (__bf16)f.w;
            }
        }
        wait_async0();
        __syncthreads();

        if (kb + 1 < EMBED / 32) {
            __builtin_prefetch(Y + (size_t)(mb * 128 + (tid >> 1)) * EMBED + (kb + 1) * 32, 0, 0);
            __builtin_prefetch(W + (size_t)((kb + 1) * 32 + (tid >> 3)) * EMBED + nb * 64, 0, 0);
        }

        FragBF afr[2], bfr[2];
        #pragma unroll
        for (int mt = 0; mt < 2; ++mt) {
            const __bf16* p = &As[(waveM * 32 + mt * 16 + lmod) * 32 + lhalf * 8];
            afr[mt].q[0] = *(const uint4*)p;
            afr[mt].q[1] = *(const uint4*)(p + 16);
        }
        #pragma unroll
        for (int nt = 0; nt < 2; ++nt) {
            const __bf16* p = &Bs[(waveN * 32 + nt * 16 + lmod) * 32 + lhalf * 8];
            bfr[nt].q[0] = *(const uint4*)p;
            bfr[nt].q[1] = *(const uint4*)(p + 16);
        }
        #pragma unroll
        for (int mt = 0; mt < 2; ++mt)
            #pragma unroll
            for (int nt = 0; nt < 2; ++nt)
                acc[mt][nt] = wmma_bf16(afr[mt], bfr[nt], acc[mt][nt]);
        __syncthreads();
    }

    #pragma unroll
    for (int nt = 0; nt < 2; ++nt) {
        const int n = nb * 64 + waveN * 32 + nt * 16 + lmod;
        const float bv = bias[n];
        #pragma unroll
        for (int mt = 0; mt < 2; ++mt) {
            FragF cf; cf.v = acc[mt][nt];
            #pragma unroll
            for (int r = 0; r < 8; ++r) {
                const int g = mb * 128 + waveM * 32 + mt * 16 + r + lhalf * 8;
                Out[(size_t)g * EMBED + n] = cf.f[r] + bv;
            }
        }
    }
}

// ---------------------------------------------------------------------------
extern "C" void kernel_launch(void* const* d_in, const int* in_sizes, int n_in,
                              void* d_out, int out_size, void* d_ws, size_t ws_size,
                              hipStream_t stream) {
    (void)in_sizes; (void)n_in; (void)out_size;
    const float* x    = (const float*)d_in[0];
    const float* Wqkv = (const float*)d_in[1];
    const float* bqkv = (const float*)d_in[2];
    const float* Wo   = (const float*)d_in[3];
    const float* bo   = (const float*)d_in[4];
    float* out = (float*)d_out;

    const size_t n_elems = (size_t)ROWS * EMBED;        // 16M elements
    const size_t need = n_elems * sizeof(__bf16) * 4;   // q, k, v, y = 128 MB
    if (ws_size < need) return;  // insufficient scratch -> no-op (no corruption)

    __bf16* q = (__bf16*)d_ws;
    __bf16* k = q + n_elems;
    __bf16* v = k + n_elems;
    __bf16* y = v + n_elems;

    qkv_gemm_kernel<<<dim3(ROWS / 128, (3 * EMBED) / 64), 256, 0, stream>>>(
        x, Wqkv, bqkv, q, k, v);
    attn_kernel<<<dim3(BATCH * HEADS), 256, 0, stream>>>(q, k, v, y);
    out_gemm_kernel<<<dim3(ROWS / 128, EMBED / 64), 256, 0, stream>>>(
        y, Wo, bo, out);
}
